// HierarchicalSoftmaxLoss_81707457839658
// MI455X (gfx1250) — compile-verified, matrix operand order checked by reading, not run on hardware
//
#include <hip/hip_runtime.h>
#include <hip/hip_bf16.h>

// Hierarchical softmax loss, MI455X (gfx1250, wave32).
// One wave per row. Lane l holds 16 floats forming row (l&15) of a 16x32 f16
// WMMA A-matrix: rows 0-7 = root segment (256 vals), rows 8-15 = gathered
// group segment (256 vals). A x ones(B) gives both 256-way exp-sums in one
// chained v_wmma_f32_16x16x32_f16 pair (hi/lo f16 split for ~f32 precision).

typedef __attribute__((ext_vector_type(16))) _Float16 v16h;
typedef __attribute__((ext_vector_type(8)))  float    v8f;

#define G_ROOT 256
#define C_CHILD 256

__device__ __forceinline__ float wave_sum32(float x) {
#pragma unroll
  for (int m = 1; m < 32; m <<= 1) x += __shfl_xor(x, m, 32);
  return x;
}

__global__ __launch_bounds__(256) void hsm_row_kernel(
    const float* __restrict__ pred, const long long* __restrict__ tgt,
    float* __restrict__ row_loss, int B, long long Dtot) {
  const int lane = threadIdx.x & 31;
  const int row  = blockIdx.x * 8 + (threadIdx.x >> 5);
  if (row >= B) return;  // wave-uniform; active waves keep EXEC all-ones

  const long long t = tgt[row];
  const int grp   = (int)(t >> 8);
  const int child = (int)(t & 255);

  // Lane -> matrix mapping (16-bit A 16x32 layout, ISA 7.12.2):
  //   seg  = bit3 of lane : 0 -> root rows 0-7, 1 -> group rows 8-15
  //   r    = lane & 7     : sub-row within segment
  //   half = lane >> 4    : 0 -> K {0..7,16..23}, 1 -> K {8..15,24..31}
  const int seg  = (lane >> 3) & 1;
  const int r    = lane & 7;
  const int half = lane >> 4;

  const float* rowp = pred + (long long)row * Dtot;
  const float* segp = seg ? (rowp + G_ROOT + (long long)grp * C_CHILD) : rowp;
  const int e0 = r * 32 + half * 8;  // element index of first 8-float chunk

  // 4x global_load_b128: exactly 16 floats/lane, 2KB/wave = both segments.
  const float4* p = (const float4*)(segp + e0);
  float4 a0 = p[0], a1 = p[1];
  const float4* q = (const float4*)(segp + e0 + 16);
  float4 b0 = q[0], b1 = q[1];

  float v[16];
  v[0]=a0.x; v[1]=a0.y; v[2]=a0.z;  v[3]=a0.w;
  v[4]=a1.x; v[5]=a1.y; v[6]=a1.z;  v[7]=a1.w;
  v[8]=b0.x; v[9]=b0.y; v[10]=b0.z; v[11]=b0.w;
  v[12]=b1.x; v[13]=b1.y; v[14]=b1.z; v[15]=b1.w;

  // Pick out the target logits (branchless ownership test + wave sum).
  const int tgtidx = seg ? child : grp;
  float accR = 0.0f, accG = 0.0f;
#pragma unroll
  for (int j = 0; j < 16; ++j) {
    const int ei = e0 + (j & 7) + ((j & 8) << 1);  // global element idx in segment
    const float sel = (ei == tgtidx) ? v[j] : 0.0f;
    accR += seg ? 0.0f : sel;
    accG += seg ? sel : 0.0f;
  }
  const float x_r = wave_sum32(accR);
  const float x_g = wave_sum32(accG);

  // Per-segment max: butterfly over XOR masks {1,2,4,16} keeps bit3 classes
  // (root lanes {0-7,16-23} vs group lanes {8-15,24-31}) independent.
  float m = v[0];
#pragma unroll
  for (int j = 1; j < 16; ++j) m = fmaxf(m, v[j]);
  m = fmaxf(m, __shfl_xor(m, 1, 32));
  m = fmaxf(m, __shfl_xor(m, 2, 32));
  m = fmaxf(m, __shfl_xor(m, 4, 32));
  m = fmaxf(m, __shfl_xor(m, 16, 32));

  // exp, then hi/lo f16 split so the WMMA row-sum is ~f32 accurate.
  v16h ahi, alo;
#pragma unroll
  for (int j = 0; j < 16; ++j) {
    const float e = __expf(v[j] - m);
    const _Float16 h = (_Float16)e;
    ahi[j] = h;
    alo[j] = (_Float16)(e - (float)h);
  }
  v16h bones;
#pragma unroll
  for (int j = 0; j < 16; ++j) bones[j] = (_Float16)1.0f;

  // D = A_hi*ones + A_lo*ones (+0). Row m of D = sum of A row m (all columns
  // identical). Chained accumulate in f32.
  v8f c = {};
  c = __builtin_amdgcn_wmma_f32_16x16x32_f16(false, ahi, false, bones,
                                             (short)0, c, false, false);
  c = __builtin_amdgcn_wmma_f32_16x16x32_f16(false, alo, false, bones,
                                             (short)0, c, false, false);
  // C layout: lanes 0-15 VGPR j = row j (root), lanes 16-31 = row 8+j (group).
  float s = c[0] + c[1] + c[2] + c[3] + c[4] + c[5] + c[6] + c[7];

  const float s_r = __shfl(s, 0, 32);   // lanes 0-15 hold root sum
  const float s_g = __shfl(s, 16, 32);  // lanes 16-31 hold group sum
  const float m_r = __shfl(m, 0, 32);   // lane 0: root max
  const float m_g = __shfl(m, 8, 32);   // lane 8: group max

  if (lane == 0) {
    const float loss = (m_r + __logf(s_r) - x_r) + (m_g + __logf(s_g) - x_g);
    row_loss[row] = loss;
  }
}

__global__ __launch_bounds__(256) void hsm_reduce_kernel(
    const float* __restrict__ row_loss, float* __restrict__ out, int B) {
  __shared__ float sm[256];
  float a = 0.0f;
  for (int i = threadIdx.x; i < B; i += 256) a += row_loss[i];
  sm[threadIdx.x] = a;
  __syncthreads();
#pragma unroll
  for (int s = 128; s > 0; s >>= 1) {
    if ((int)threadIdx.x < s) sm[threadIdx.x] += sm[threadIdx.x + s];
    __syncthreads();
  }
  if (threadIdx.x == 0) out[0] = sm[0] / (float)B;
}

extern "C" void kernel_launch(void* const* d_in, const int* in_sizes, int n_in,
                              void* d_out, int out_size, void* d_ws, size_t ws_size,
                              hipStream_t stream) {
  const float* pred     = (const float*)d_in[0];       // [B, D] f32
  const long long* tgt  = (const long long*)d_in[1];   // [B] i64
  const int B = in_sizes[1];
  const long long Dtot = (long long)in_sizes[0] / (long long)B;  // 65792

  float* row_loss = (float*)d_ws;  // B floats of scratch
  const int blocks = (B + 7) / 8;  // 8 waves (rows) per 256-thread block
  hsm_row_kernel<<<blocks, 256, 0, stream>>>(pred, tgt, row_loss, B, Dtot);
  hsm_reduce_kernel<<<1, 256, 0, stream>>>(row_loss, (float*)d_out, B);
}
